// MOE_MHC_41180146434711
// MI455X (gfx1250) — matrix operand order repaired and checked by reading, not compiled, not run on hardware
//
#include <hip/hip_runtime.h>

// ---------------------------------------------------------------------------
// MoE-transformer forward for MI455X (gfx1250).
// Heavy GEMMs run on v_wmma_f32_16x16x32_bf16 (fp32 accumulate, bf16 operands).
// All GEMM operands are padded so M%64==0, N%64==0, K%32==0 -> guard-free
// inner loop. Requires ~1.08 GB of d_ws scratch.
// ---------------------------------------------------------------------------

#define D_MODEL   512
#define N_HEADS   8
#define D_HEAD    64
#define N_EXPERTS 16
#define BATCH     1024
#define S_ALL     76
#define CAPSLOTS  19
#define NROWS     (BATCH * S_ALL)   /* 77824 token rows */

#define TM 64
#define TN 64
#define TKDEPTH 32

// classifier padded dims
#define H1PAD 1024   /* 1000 -> 1024 */
#define H2PAD 128    /* 100  -> 128  */

typedef __attribute__((ext_vector_type(16))) __bf16         v16bf;
typedef __attribute__((ext_vector_type(16))) unsigned short v16u;
typedef __attribute__((ext_vector_type(8)))  float          v8f;

static __device__ __forceinline__ unsigned short f32_to_bf16_rne(float f) {
  unsigned int u = __builtin_bit_cast(unsigned int, f);
  u += 0x7fffu + ((u >> 16) & 1u);            // round-to-nearest-even
  return (unsigned short)(u >> 16);
}
static __device__ __forceinline__ unsigned int pack2_bf16(float a, float b) {
  return (unsigned int)f32_to_bf16_rne(a) | ((unsigned int)f32_to_bf16_rne(b) << 16);
}

// ---------------------------------------------------------------------------
// Guard-free bf16 WMMA GEMM:  C[M,N](f32) = mapA(A)[M,K](bf16) * B[K,N](bf16)
// PRECONDITIONS: M%64==0, N%64==0, K%32==0; A rows 16B-aligned (lda%8==0,
// base 256B-aligned); B rows 16B-aligned; for gather mode zrow has >= K
// zeroed bf16 entries.
//   row map (no gather): phys = (m / sBatch) * rowStride + (m % sBatch) + rowOff
//   gather:              phys = gatherIdx[m] (-1 => zrow), C row = m
//   epilogue: optional bias[n], act==1 -> ReLU
// ---------------------------------------------------------------------------
__global__ __launch_bounds__(128) void gemm_bf16_wmma_kernel(
    const unsigned short* __restrict__ A, const unsigned short* __restrict__ Bw,
    float* __restrict__ C,
    int K, int lda, int ldb, int ldc,
    int sBatch, int rowOff, int rowStride,
    const int* __restrict__ gatherIdx,
    const unsigned short* __restrict__ zrow,
    const float* __restrict__ bias, int act)
{
  __shared__ unsigned short As[TM * TKDEPTH];   // [row][k]
  __shared__ unsigned short Bs[TN * TKDEPTH];   // [n][k]  (transposed stage)

  const int t    = threadIdx.x;
  const int gm   = blockIdx.y * TM;
  const int gn   = blockIdx.x * TN;
  const int wave = t >> 5;
  const int lane = t & 31;
  const int half = lane >> 4;
  const int l16  = lane & 15;

  v8f acc[4];
#pragma unroll
  for (int nb = 0; nb < 4; ++nb)
#pragma unroll
    for (int v = 0; v < 8; ++v) acc[nb][v] = 0.0f;

  // A-stage coordinates: thread fills 16 contiguous k's of one tile row.
  const int ar  = t >> 1;
  const int ac0 = (t & 1) * 16;
  const int am  = gm + ar;
  long long arow;
  if (gatherIdx) arow = (long long)gatherIdx[am];
  else           arow = (long long)(am / sBatch) * rowStride + (am % sBatch) + rowOff;
  const unsigned short* Aptr = (arow >= 0) ? (A + arow * (long long)lda) : zrow;

  // B-stage coordinates: thread handles k-pair kp (rows 2kp,2kp+1) x 8 n's.
  const int kp  = t >> 3;
  const int nb0 = (t & 7) * 8;
  const unsigned short* Bptr = Bw + (long long)gn + nb0;

  for (int k0 = 0; k0 < K; k0 += TKDEPTH) {
    if (k0 + TKDEPTH < K)
      __builtin_prefetch(Aptr + k0 + TKDEPTH, 0, 1);   // global_prefetch_b8

    // ---- stage A tile (64 x 32): 2x b128 load, 2x b128 LDS store ----
    {
      const uint4* ap = (const uint4*)(Aptr + k0 + ac0);
      const uint4 a0 = ap[0];
      const uint4 a1 = ap[1];
      uint4* dst = (uint4*)&As[ar * TKDEPTH + ac0];
      dst[0] = a0;
      dst[1] = a1;
    }
    // ---- stage B tile (32 x 64) -> Bs[n][k], k-pairs packed as b32 ----
    {
      const unsigned short* bp = Bptr + (long long)(k0 + 2 * kp) * ldb;
      const uint4 r0 = *(const uint4*)bp;
      const uint4 r1 = *(const uint4*)(bp + ldb);
      const unsigned int w0[4] = {r0.x, r0.y, r0.z, r0.w};
      const unsigned int w1[4] = {r1.x, r1.y, r1.z, r1.w};
#pragma unroll
      for (int j = 0; j < 8; ++j) {
        const unsigned int lo = (w0[j >> 1] >> ((j & 1) * 16)) & 0xffffu;
        const unsigned int hi = (w1[j >> 1] >> ((j & 1) * 16)) & 0xffffu;
        *(unsigned int*)&Bs[(nb0 + j) * TKDEPTH + 2 * kp] = lo | (hi << 16);
      }
    }
    __syncthreads();

    // A fragment 16x32 bf16: lane L<16 -> M=L; VGPR j<4: K=half*8+2j(+1);
    // j>=4: K=16+half*8+2(j-4)(+1).
    v16u au;
    const int amrow = wave * 16 + l16;
#pragma unroll
    for (int j = 0; j < 8; ++j) {
      const int kb = (j < 4) ? (half * 8 + 2 * j) : (16 + half * 8 + 2 * (j - 4));
      au[2 * j]     = As[amrow * TKDEPTH + kb];
      au[2 * j + 1] = As[amrow * TKDEPTH + kb + 1];
    }
    const v16bf afrag = __builtin_bit_cast(v16bf, au);

#pragma unroll
    for (int nb = 0; nb < 4; ++nb) {
      // B fragment 32x16 bf16: lane gives N=l16; element idx<8 -> K=half*8+idx, idx>=8 -> K=16+half*8+(idx-8).
      v16u bu;
      const int ncol = nb * 16 + l16;
#pragma unroll
      for (int idx = 0; idx < 16; ++idx)
        bu[idx] = Bs[ncol * TKDEPTH + ((idx < 8) ? (half * 8 + idx) : (16 + half * 8 + (idx - 8)))];
      const v16bf bfrag = __builtin_bit_cast(v16bf, bu);
      acc[nb] = __builtin_amdgcn_wmma_f32_16x16x32_bf16(
          false, afrag, false, bfrag, (short)0, acc[nb], false, false);
    }
    __syncthreads();
  }

  // Epilogue. C 16x16 f32 layout: VGPR v, lanes0-15 M=v, lanes16-31 M=8+v, N=l16.
#pragma unroll
  for (int nb = 0; nb < 4; ++nb) {
#pragma unroll
    for (int v = 0; v < 8; ++v) {
      const int mrow = gm + wave * 16 + half * 8 + v;
      const int ncol = gn + nb * 16 + l16;
      float val = acc[nb][v];
      if (bias) val += bias[ncol];
      if (act == 1) val = fmaxf(val, 0.0f);
      const long long crow = gatherIdx
          ? (long long)mrow
          : (long long)(mrow / sBatch) * rowStride + (mrow % sBatch) + rowOff;
      C[crow * (long long)ldc + ncol] = val;
    }
  }
}

// ---------------------------------------------------------------------------
// Embedding gather + sinusoidal PE.  x row index = b*76 + rowOff + s.
// ---------------------------------------------------------------------------
__global__ void embed_pe_kernel(const int* __restrict__ tok,
                                const float* __restrict__ Wemb,
                                float* __restrict__ x, int S, int rowOff)
{
  const long long i = (long long)blockIdx.x * blockDim.x + threadIdx.x;
  const long long total = (long long)BATCH * S * D_MODEL;
  if (i >= total) return;
  const int d = (int)(i & (D_MODEL - 1));
  const long long bs = i >> 9;
  const int s = (int)(bs % S);
  const int b = (int)(bs / S);
  const int tk = tok[b * S + s];
  const int even = (d >> 1) << 1;
  const float div = __expf(-(float)even * (9.210340371976184f / (float)D_MODEL));
  const float ang = (float)s * div;
  const float pe = (d & 1) ? __cosf(ang) : __sinf(ang);
  x[((long long)b * S_ALL + rowOff + s) * D_MODEL + d] = Wemb[tk * D_MODEL + d] + pe;
}

// Vectorized converters: 4 floats -> 4 bf16 per thread.
__global__ void f32_to_bf16_kernel(const float* __restrict__ in,
                                   unsigned short* __restrict__ out, long long n)
{
  const long long i4 = (long long)blockIdx.x * blockDim.x + threadIdx.x;
  const long long i = i4 * 4;
  if (i + 3 < n) {
    const float4 f = ((const float4*)in)[i4];
    uint2 r;
    r.x = pack2_bf16(f.x, f.y);
    r.y = pack2_bf16(f.z, f.w);
    ((uint2*)out)[i4] = r;
  } else {
    for (long long j = i; j < n; ++j) out[j] = f32_to_bf16_rne(in[j]);
  }
}

// f32 [rows x cols] -> bf16 [rowsPad x colsPad], zero padding.
__global__ void f32_to_bf16_pad_kernel(const float* __restrict__ in,
                                       unsigned short* __restrict__ out,
                                       int rows, int cols, int rowsPad, int colsPad)
{
  const long long i = (long long)blockIdx.x * blockDim.x + threadIdx.x;
  const long long total = (long long)rowsPad * colsPad;
  if (i >= total) return;
  const int c = (int)(i % colsPad);
  const long long r = i / colsPad;
  const float v = (r < rows && c < cols) ? in[r * cols + c] : 0.0f;
  out[i] = f32_to_bf16_rne(v);
}

__global__ void pad_bias_kernel(const float* __restrict__ in, float* __restrict__ out,
                                int n, int npad)
{
  const int i = blockIdx.x * blockDim.x + threadIdx.x;
  if (i < npad) out[i] = (i < n) ? in[i] : 0.0f;
}

__global__ void zero_u16_kernel(unsigned short* __restrict__ p, int n)
{
  const int i = blockIdx.x * blockDim.x + threadIdx.x;
  if (i < n) p[i] = 0;
}

static __device__ __forceinline__ float gelu_tanh(float v) {
  return 0.5f * v * (1.0f + tanhf(0.7978845608028654f * (v + 0.044715f * v * v * v)));
}

__global__ void gelu_to_bf16_kernel(const float* __restrict__ in,
                                    unsigned short* __restrict__ out, long long n)
{
  const long long i4 = (long long)blockIdx.x * blockDim.x + threadIdx.x;
  const long long i = i4 * 4;
  if (i + 3 < n) {
    const float4 f = ((const float4*)in)[i4];
    uint2 r;
    r.x = pack2_bf16(gelu_tanh(f.x), gelu_tanh(f.y));
    r.y = pack2_bf16(gelu_tanh(f.z), gelu_tanh(f.w));
    ((uint2*)out)[i4] = r;
  } else {
    for (long long j = i; j < n; ++j) out[j] = f32_to_bf16_rne(gelu_tanh(in[j]));
  }
}

__global__ void copy_f32_kernel(float* __restrict__ dst,
                                const float* __restrict__ src, long long n)
{
  const long long i4 = (long long)blockIdx.x * blockDim.x + threadIdx.x;
  const long long i = i4 * 4;
  if (i + 3 < n) {
    ((float4*)dst)[i4] = ((const float4*)src)[i4];
  } else {
    for (long long j = i; j < n; ++j) dst[j] = src[j];
  }
}

// ---------------------------------------------------------------------------
// Self-attention for one (b, h): S<=76, dh=64. K/V staged in LDS.
// ---------------------------------------------------------------------------
__global__ __launch_bounds__(128) void attn_kernel(
    const float* __restrict__ q, const float* __restrict__ k,
    const float* __restrict__ v, float* __restrict__ o, int S, int rowOff)
{
  __shared__ float kL[S_ALL * D_HEAD];
  __shared__ float vL[S_ALL * D_HEAD];
  const int bh = blockIdx.x;
  const int b = bh / N_HEADS, h = bh % N_HEADS;
  const long long base = ((long long)b * S_ALL + rowOff) * D_MODEL + h * D_HEAD;

  for (int i = threadIdx.x; i < S * D_HEAD; i += blockDim.x) {
    const int s = i >> 6, d = i & 63;
    kL[i] = k[base + (long long)s * D_MODEL + d];
    vL[i] = v[base + (long long)s * D_MODEL + d];
  }
  __syncthreads();

  const int s = threadIdx.x;
  if (s >= S) return;

  float qreg[D_HEAD];
#pragma unroll
  for (int d = 0; d < D_HEAD; ++d) qreg[d] = q[base + (long long)s * D_MODEL + d];

  float sc[S_ALL];
  float mx = -1e30f;
  for (int tt = 0; tt < S; ++tt) {
    float dot = 0.f;
#pragma unroll
    for (int d = 0; d < D_HEAD; ++d) dot += qreg[d] * kL[tt * D_HEAD + d];
    dot *= 0.125f;                     // 1/sqrt(64)
    sc[tt] = dot;
    mx = fmaxf(mx, dot);
  }
  float sum = 0.f;
  for (int tt = 0; tt < S; ++tt) { const float e = __expf(sc[tt] - mx); sc[tt] = e; sum += e; }
  const float inv = 1.0f / sum;

  float outv[D_HEAD];
#pragma unroll
  for (int d = 0; d < D_HEAD; ++d) outv[d] = 0.f;
  for (int tt = 0; tt < S; ++tt) {
    const float w = sc[tt] * inv;
#pragma unroll
    for (int d = 0; d < D_HEAD; ++d) outv[d] += w * vL[tt * D_HEAD + d];
  }
#pragma unroll
  for (int d = 0; d < D_HEAD; ++d) o[base + (long long)s * D_MODEL + d] = outv[d];
}

// ---------------------------------------------------------------------------
// RMSNorm (optional residual) over last dim n; one block per logical row.
// ---------------------------------------------------------------------------
__global__ __launch_bounds__(256) void rmsnorm_kernel(
    float* __restrict__ x, const float* __restrict__ res, const float* __restrict__ w,
    int n, int ld, int sBatch, int rowOff, int rowStride)
{
  const int m = blockIdx.x;
  const long long row = (long long)(m / sBatch) * rowStride + (m % sBatch) + rowOff;
  float* xr = x + row * ld;
  const float* rr = res ? res + row * ld : nullptr;

  __shared__ float red[256];
  float ss = 0.f;
  for (int i = threadIdx.x; i < n; i += 256) {
    const float vv = xr[i] + (rr ? rr[i] : 0.f);
    ss += vv * vv;
  }
  red[threadIdx.x] = ss;
  __syncthreads();
  for (int st = 128; st > 0; st >>= 1) {
    if (threadIdx.x < st) red[threadIdx.x] += red[threadIdx.x + st];
    __syncthreads();
  }
  const float inv = rsqrtf(red[0] / (float)n + 1e-6f);
  for (int i = threadIdx.x; i < n; i += 256) {
    const float vv = xr[i] + (rr ? rr[i] : 0.f);
    xr[i] = vv * inv * w[i];
  }
}

// ---------------------------------------------------------------------------
// MoE gating: softmax over 16 experts, top-2, normalized gates.
// ---------------------------------------------------------------------------
__global__ void moe_gate_kernel(const float* __restrict__ x, const float* __restrict__ gw,
                                int* __restrict__ i1, int* __restrict__ i2,
                                float* __restrict__ g1, float* __restrict__ g2, int T)
{
  const int t = blockIdx.x * blockDim.x + threadIdx.x;
  if (t >= T) return;
  const float* xr = x + (long long)t * D_MODEL;
  float logit[N_EXPERTS];
#pragma unroll
  for (int e = 0; e < N_EXPERTS; ++e) logit[e] = 0.f;
  for (int d = 0; d < D_MODEL; ++d) {
    const float xv = xr[d];
#pragma unroll
    for (int e = 0; e < N_EXPERTS; ++e) logit[e] += xv * gw[d * N_EXPERTS + e];
  }
  float mx = logit[0];
#pragma unroll
  for (int e = 1; e < N_EXPERTS; ++e) mx = fmaxf(mx, logit[e]);
  float p[N_EXPERTS]; float sum = 0.f;
#pragma unroll
  for (int e = 0; e < N_EXPERTS; ++e) { p[e] = __expf(logit[e] - mx); sum += p[e]; }
  const float inv = 1.0f / sum;
#pragma unroll
  for (int e = 0; e < N_EXPERTS; ++e) p[e] *= inv;

  int a1 = 0; float b1v = p[0];
#pragma unroll
  for (int e = 1; e < N_EXPERTS; ++e) if (p[e] > b1v) { b1v = p[e]; a1 = e; }
  int a2 = -1; float b2v = -1.f;
#pragma unroll
  for (int e = 0; e < N_EXPERTS; ++e) if (e != a1 && p[e] > b2v) { b2v = p[e]; a2 = e; }
  const float den = b1v + b2v + 1e-9f;
  i1[t] = a1; i2[t] = a2; g1[t] = b1v / den; g2[t] = b2v / den;
}

__global__ void init_slots_kernel(int* __restrict__ src, float* __restrict__ gate, int n)
{
  const int i = blockIdx.x * blockDim.x + threadIdx.x;
  if (i < n) { src[i] = -1; gate[i] = 0.f; }
}

// Serial capacity pass per batch row (matches cumsum-based reference exactly).
__global__ void moe_capacity_kernel(const int* __restrict__ i1, const int* __restrict__ i2,
                                    const float* __restrict__ g1, const float* __restrict__ g2,
                                    int* __restrict__ slot_src, float* __restrict__ slot_gate)
{
  const int b = blockIdx.x * blockDim.x + threadIdx.x;
  if (b >= BATCH) return;
  int cnt1[N_EXPERTS], c1c[N_EXPERTS], cnt2[N_EXPERTS];
  for (int e = 0; e < N_EXPERTS; ++e) { cnt1[e] = 0; c1c[e] = 0; cnt2[e] = 0; }
  for (int s = 0; s < S_ALL; ++s) {
    const int t = b * S_ALL + s;
    const int e = i1[t];
    const int p = cnt1[e]++;
    if (p < CAPSLOTS) {
      const long long sl = ((long long)e * BATCH + b) * CAPSLOTS + p;
      slot_src[sl] = t; slot_gate[sl] = g1[t]; c1c[e]++;
    }
  }
  for (int s = 0; s < S_ALL; ++s) {
    const int t = b * S_ALL + s;
    const int e = i2[t];
    const int p = c1c[e] + cnt2[e]++;
    if (p < CAPSLOTS) {
      const long long sl = ((long long)e * BATCH + b) * CAPSLOTS + p;
      slot_src[sl] = t; slot_gate[sl] = g2[t];
    }
  }
}

// y[src_token, :] += gate * eo[slot, :]   (one expert per launch -> no atomics)
__global__ void moe_scatter_kernel(const float* __restrict__ eo,
                                   const int* __restrict__ src,
                                   const float* __restrict__ gate,
                                   float* __restrict__ y, int nslots)
{
  const long long i = (long long)blockIdx.x * blockDim.x + threadIdx.x;
  const long long total = (long long)nslots * D_MODEL;
  if (i >= total) return;
  const int r = (int)(i >> 9);
  const int d = (int)(i & (D_MODEL - 1));
  const int sidx = src[r];
  if (sidx >= 0) y[(long long)sidx * D_MODEL + d] += gate[r] * eo[i];
}

__global__ void final_head_kernel(const float* __restrict__ h2, const float* __restrict__ w3,
                                  const float* __restrict__ b3, float* __restrict__ out,
                                  int n, int ld)
{
  const int b = blockIdx.x * blockDim.x + threadIdx.x;
  if (b >= BATCH) return;
  float s = 0.f;
  for (int j = 0; j < n; ++j) s += h2[(long long)b * ld + j] * w3[j];
  out[b] = s + b3[0];
}

// ---------------------------------------------------------------------------
extern "C" void kernel_launch(void* const* d_in, const int* in_sizes, int n_in,
                              void* d_out, int out_size, void* d_ws, size_t ws_size,
                              hipStream_t stream)
{
  (void)in_sizes; (void)n_in; (void)out_size; (void)ws_size;
  const int*   tokA  = (const int*)d_in[0];
  const int*   tokB  = (const int*)d_in[1];
  const int*   tokP  = (const int*)d_in[2];
  const float* WembA = (const float*)d_in[3];
  const float* WembB = (const float*)d_in[4];
  const float* WembP = (const float*)d_in[5];
  const float* attnW = (const float*)d_in[6];
  const float* rmsD  = (const float*)d_in[7];
  const float* gateW = (const float*)d_in[8];
  const float* eW1   = (const float*)d_in[9];
  const float* eW2   = (const float*)d_in[10];
  const float* cW1   = (const float*)d_in[11];
  const float* cB1   = (const float*)d_in[12];
  const float* rms1  = (const float*)d_in[13];
  const float* cW2   = (const float*)d_in[14];
  const float* cB2   = (const float*)d_in[15];
  const float* rms2  = (const float*)d_in[16];
  const float* cW3   = (const float*)d_in[17];
  const float* cB3   = (const float*)d_in[18];
  float* out = (float*)d_out;

  const long long XE = (long long)NROWS * D_MODEL;   // 39,845,888 elems
  const long long DD = (long long)D_MODEL * D_MODEL; // 262,144
  const int MOE_M = BATCH * CAPSLOTS;                // 19456
  const int FDIM  = S_ALL * D_MODEL;                 // 38912

  char* base = (char*)d_ws;
  size_t off = 0;
  auto alloc = [&](size_t bytes) -> char* {
    off = (off + 255) & ~(size_t)255;
    char* p = base + off; off += bytes; return p;
  };

  float* x  = (float*)alloc(XE * 4);
  float* qf = (float*)alloc(XE * 4);   // also proj-tmp
  float* kf = (float*)alloc(XE * 4);   // also y (MoE output + residual)
  float* vf = (float*)alloc(XE * 4);   // also hid, then h1 [1024 x H1PAD]
  float* of = (float*)alloc(XE * 4);   // also eo
  unsigned short* xb  = (unsigned short*)alloc(XE * 2);   // bf16 activations / fb
  unsigned short* obh = (unsigned short*)alloc(XE * 2);   // bf16 attn-out / hidb / h1b
  unsigned short* wAb  = (unsigned short*)alloc(16 * DD * 2);
  unsigned short* wE1b = (unsigned short*)alloc(N_EXPERTS * DD * 2);
  unsigned short* wE2b = (unsigned short*)alloc(N_EXPERTS * DD * 2);
  unsigned short* wC1b = (unsigned short*)alloc((size_t)FDIM * H1PAD * 2);  // padded N
  unsigned short* wC2b = (unsigned short*)alloc((size_t)H1PAD * H2PAD * 2); // padded K,N
  float* b1p = (float*)alloc(H1PAD * 4);
  float* b2p = (float*)alloc(H2PAD * 4);
  unsigned short* zrow = (unsigned short*)alloc(D_MODEL * 2);  // zero A-row for gather
  int*   i1v = (int*)alloc((size_t)NROWS * 4);
  int*   i2v = (int*)alloc((size_t)NROWS * 4);
  float* g1v = (float*)alloc((size_t)NROWS * 4);
  float* g2v = (float*)alloc((size_t)NROWS * 4);
  int*   slot_src  = (int*)alloc((size_t)N_EXPERTS * BATCH * CAPSLOTS * 4);
  float* slot_gate = (float*)alloc((size_t)N_EXPERTS * BATCH * CAPSLOTS * 4);
  float* h2 = (float*)alloc((size_t)BATCH * H2PAD * 4);
  float* y = kf; float* hid = vf; float* eo = of;
  unsigned short* hidb = obh;
  float* h1 = vf; unsigned short* h1b = obh;

  auto conv = [&](const float* in, unsigned short* o, long long n) {
    const long long n4 = (n + 3) / 4;
    f32_to_bf16_kernel<<<dim3((unsigned)((n4 + 255) / 256)), 256, 0, stream>>>(in, o, n);
  };
  auto gemm = [&](const unsigned short* A, const unsigned short* Bw, float* C,
                  int M, int N, int K, int lda, int ldb, int ldc,
                  int sBatch, int rowOff, int rowStride,
                  const int* gidx, const float* bias, int act) {
    dim3 g((unsigned)(N / TN), (unsigned)(M / TM));
    gemm_bf16_wmma_kernel<<<g, 128, 0, stream>>>(A, Bw, C, K, lda, ldb, ldc,
                                                 sBatch, rowOff, rowStride, gidx, zrow,
                                                 bias, act);
  };

  // --- weights -> bf16 (classifier weights padded) ---
  conv(attnW, wAb, 16 * DD);
  conv(eW1, wE1b, N_EXPERTS * DD);
  conv(eW2, wE2b, N_EXPERTS * DD);
  {
    long long n = (long long)FDIM * H1PAD;
    f32_to_bf16_pad_kernel<<<dim3((unsigned)((n + 255) / 256)), 256, 0, stream>>>(
        cW1, wC1b, FDIM, 1000, FDIM, H1PAD);
    n = (long long)H1PAD * H2PAD;
    f32_to_bf16_pad_kernel<<<dim3((unsigned)((n + 255) / 256)), 256, 0, stream>>>(
        cW2, wC2b, 1000, 100, H1PAD, H2PAD);
  }
  pad_bias_kernel<<<(H1PAD + 255) / 256, 256, 0, stream>>>(cB1, b1p, 1000, H1PAD);
  pad_bias_kernel<<<(H2PAD + 255) / 256, 256, 0, stream>>>(cB2, b2p, 100, H2PAD);
  zero_u16_kernel<<<(D_MODEL + 255) / 256, 256, 0, stream>>>(zrow, D_MODEL);

  // --- embeddings + PE ---
  {
    long long n = (long long)BATCH * 34 * D_MODEL;
    embed_pe_kernel<<<dim3((unsigned)((n + 255) / 256)), 256, 0, stream>>>(tokA, WembA, x, 34, 0);
    n = (long long)BATCH * 28 * D_MODEL;
    embed_pe_kernel<<<dim3((unsigned)((n + 255) / 256)), 256, 0, stream>>>(tokB, WembB, x, 28, 34);
    n = (long long)BATCH * 14 * D_MODEL;
    embed_pe_kernel<<<dim3((unsigned)((n + 255) / 256)), 256, 0, stream>>>(tokP, WembP, x, 14, 62);
  }
  conv(x, xb, XE);

  // --- 4 attention encoders (a, b, p on disjoint row slices, then joint) ---
  const int Ss[4]   = {34, 28, 14, S_ALL};
  const int offs[4] = {0, 34, 62, 0};
  for (int enc = 0; enc < 4; ++enc) {
    if (enc == 3) conv(x, xb, XE);     // joint encoder needs post-branch activations
    const int S = Ss[enc], ro = offs[enc];
    const int M = BATCH * S;           // all multiples of 64
    const unsigned short* wq = wAb + (size_t)(enc * 4 + 0) * DD;
    const unsigned short* wk = wAb + (size_t)(enc * 4 + 1) * DD;
    const unsigned short* wv = wAb + (size_t)(enc * 4 + 2) * DD;
    const unsigned short* wo = wAb + (size_t)(enc * 4 + 3) * DD;
    gemm(xb, wq, qf, M, D_MODEL, D_MODEL, D_MODEL, D_MODEL, D_MODEL, S, ro, S_ALL, nullptr, nullptr, 0);
    gemm(xb, wk, kf, M, D_MODEL, D_MODEL, D_MODEL, D_MODEL, D_MODEL, S, ro, S_ALL, nullptr, nullptr, 0);
    gemm(xb, wv, vf, M, D_MODEL, D_MODEL, D_MODEL, D_MODEL, D_MODEL, S, ro, S_ALL, nullptr, nullptr, 0);
    attn_kernel<<<BATCH * N_HEADS, 128, 0, stream>>>(qf, kf, vf, of, S, ro);
    conv(of, obh, XE);                 // full-buffer convert; only mapped rows consumed
    gemm(obh, wo, qf, M, D_MODEL, D_MODEL, D_MODEL, D_MODEL, D_MODEL, S, ro, S_ALL, nullptr, nullptr, 0);
    rmsnorm_kernel<<<M, 256, 0, stream>>>(x, qf, rmsD, D_MODEL, D_MODEL, S, ro, S_ALL);
  }
  conv(x, xb, XE);                     // bf16 of post-encoder activations for MoE dispatch

  // --- MoE: top-2 gating, capacity, 16 gather-GEMM expert chains ---
  moe_gate_kernel<<<(NROWS + 255) / 256, 256, 0, stream>>>(x, gateW, i1v, i2v, g1v, g2v, NROWS);
  {
    const int nslots = N_EXPERTS * BATCH * CAPSLOTS;
    init_slots_kernel<<<(nslots + 255) / 256, 256, 0, stream>>>(slot_src, slot_gate, nslots);
  }
  moe_capacity_kernel<<<(BATCH + 127) / 128, 128, 0, stream>>>(i1v, i2v, g1v, g2v, slot_src, slot_gate);
  copy_f32_kernel<<<dim3((unsigned)(((XE + 3) / 4 + 255) / 256)), 256, 0, stream>>>(y, x, XE);

  for (int e = 0; e < N_EXPERTS; ++e) {
    const int*   gidx  = slot_src  + (size_t)e * BATCH * CAPSLOTS;
    const float* ggate = slot_gate + (size_t)e * BATCH * CAPSLOTS;
    gemm(xb, wE1b + (size_t)e * DD, hid, MOE_M, D_MODEL, D_MODEL,
         D_MODEL, D_MODEL, D_MODEL, MOE_M, 0, 0, gidx, nullptr, 0);
    {
      const long long n = (long long)MOE_M * D_MODEL;
      const long long n4 = (n + 3) / 4;
      gelu_to_bf16_kernel<<<dim3((unsigned)((n4 + 255) / 256)), 256, 0, stream>>>(hid, hidb, n);
    }
    gemm(hidb, wE2b + (size_t)e * DD, eo, MOE_M, D_MODEL, D_MODEL,
         D_MODEL, D_MODEL, D_MODEL, MOE_M, 0, 0, nullptr, nullptr, 0);
    {
      const long long n = (long long)MOE_M * D_MODEL;
      moe_scatter_kernel<<<dim3((unsigned)((n + 255) / 256)), 256, 0, stream>>>(eo, gidx, ggate, y, MOE_M);
    }
  }

  // --- classifier head (padded to H1PAD/H2PAD; pads are exact zeros) ---
  conv(y, xb, XE);                     // fb = bf16(y) viewed as [1024, 38912]
  gemm(xb, wC1b, h1, BATCH, H1PAD, FDIM,
       FDIM, H1PAD, H1PAD, BATCH, 0, 0, nullptr, b1p, 1);
  rmsnorm_kernel<<<BATCH, 256, 0, stream>>>(h1, nullptr, rms1, 1000, H1PAD, BATCH, 0, 0);
  conv(h1, h1b, (long long)BATCH * H1PAD);
  gemm(h1b, wC2b, h2, BATCH, H2PAD, H1PAD,
       H1PAD, H2PAD, H2PAD, BATCH, 0, 0, nullptr, b2p, 1);
  rmsnorm_kernel<<<BATCH, 256, 0, stream>>>(h2, nullptr, rms2, 100, H2PAD, BATCH, 0, 0);
  final_head_kernel<<<(BATCH + 255) / 256, 256, 0, stream>>>(h2, cW3, cB3, out, 100, H2PAD);
}
